// QuantumKernel_65481071410173
// MI455X (gfx1250) — matrix operand order repaired and checked by reading, not compiled, not run on hardware
//
#include <hip/hip_runtime.h>

// ---------------------------------------------------------------------------
// RBF Gram matrix on MI455X (gfx1250, wave32, WMMA).
//   K[i,j] = exp(-gamma * max(||a_i||^2 + ||b_j||^2 - 2 a_i.b_j, 0))
// Cross term via v_wmma_f32_16x16x32_f16 (f32 accumulate); norms + exp in f32.
// Round 2: 256x128 workgroup tile, 64x64 per wave (4x4 WMMA tiles) to double
// WMMA density per LDS fragment load and halve barrier/wait overhead per WMMA.
// ---------------------------------------------------------------------------

typedef _Float16 v8h  __attribute__((ext_vector_type(8)));
typedef _Float16 v16h __attribute__((ext_vector_type(16)));
typedef float    v8f  __attribute__((ext_vector_type(8)));

#define NROWS 8192
#define KDIM  256
#define GAMMA 0.00390625f
#define BM    256          // workgroup tile rows (A)
#define BN    128          // workgroup tile cols (B)
#define KSTEP 32
#define LDSS  40           // LDS row stride in halfs: 32 + 8 pad -> 80B rows

// ---------------------------------------------------------------------------
// Prep: convert f32 rows -> f16 workspace, compute exact f32 row norms.
// One 256-thread block per row; blocks [0,8192) -> A, [8192,16384) -> B.
// ---------------------------------------------------------------------------
__global__ __launch_bounds__(256)
void rbf_prep(const float* __restrict__ a, const float* __restrict__ b,
              _Float16* __restrict__ aH, _Float16* __restrict__ bH,
              float* __restrict__ aSq, float* __restrict__ bSq)
{
    __shared__ float red[256];
    int r = blockIdx.x;
    const float* src;
    _Float16*    dst;
    float*       sq;
    int row;
    if (r < NROWS) { row = r;         src = a + (size_t)row * KDIM; dst = aH + (size_t)row * KDIM; sq = aSq; }
    else           { row = r - NROWS; src = b + (size_t)row * KDIM; dst = bH + (size_t)row * KDIM; sq = bSq; }

    int t = threadIdx.x;                 // KDIM == blockDim.x == 256
    float x = src[t];
    dst[t] = (_Float16)x;
    red[t] = x * x;
    __syncthreads();
    for (int s = 128; s > 0; s >>= 1) {
        if (t < s) red[t] += red[t + s];
        __syncthreads();
    }
    if (t == 0) sq[row] = red[0];
}

// ---------------------------------------------------------------------------
// Main kernel: 256x128 output tile per 256-thread block (8 waves, 4x2 wave
// grid, each wave computes 4x4 tiles of 16x16 = 64x64). K staged via LDS.
// ---------------------------------------------------------------------------
__global__ __launch_bounds__(256)
void rbf_wmma(const _Float16* __restrict__ aH, const _Float16* __restrict__ bH,
              const float* __restrict__ aSq, const float* __restrict__ bSq,
              float* __restrict__ out)
{
    __shared__ _Float16 lA[BM * LDSS];   // 20 KB
    __shared__ _Float16 lB[BN * LDSS];   // 10 KB

    const int tid   = threadIdx.x;
    const int lane  = tid & 31;
    const int wave  = tid >> 5;        // 0..7
    const int waveM = wave >> 1;       // 0..3 : 64 rows each
    const int waveN = wave & 1;        // 0..1 : 64 cols each
    const int lrow  = lane & 15;       // row-in-tile (A) / col-in-tile (B,C,D)
    const int lhi   = lane >> 4;       // half-wave select

    const int tileM = blockIdx.y * BM;
    const int tileN = blockIdx.x * BN;

    v8f acc[4][4];
    #pragma unroll
    for (int mt = 0; mt < 4; ++mt)
        #pragma unroll
        for (int nt = 0; nt < 4; ++nt)
            #pragma unroll
            for (int e = 0; e < 8; ++e)
                acc[mt][nt][e] = 0.0f;

    for (int kk = 0; kk < KDIM; kk += KSTEP) {
        __syncthreads();   // protect previous iteration's fragment reads
        // Stage A (256 rows x 32 halfs) and B (128 rows x 32 halfs) into LDS.
        // Chunks of 8 halfs (16B); coalesced global_load_b128 / ds_store_b128.
        #pragma unroll
        for (int i = 0; i < 4; ++i) {
            int c   = tid + i * 256;       // 0..1023
            int row = c >> 2;
            int col = (c & 3) * 8;
            *(v8h*)&lA[row * LDSS + col] =
                *(const v8h*)&aH[(size_t)(tileM + row) * KDIM + kk + col];
        }
        #pragma unroll
        for (int i = 0; i < 2; ++i) {
            int c   = tid + i * 256;       // 0..511
            int row = c >> 2;
            int col = (c & 3) * 8;
            *(v8h*)&lB[row * LDSS + col] =
                *(const v8h*)&bH[(size_t)(tileN + row) * KDIM + kk + col];
        }
        __syncthreads();

        // Build fragments per ISA 16-bit layouts (wave32).
        // A 16x32: lanes<16 hold row=lrow, K {0..7, 16..23}; lanes>=16 K {8..15, 24..31}
        v16h af[4];
        #pragma unroll
        for (int mt = 0; mt < 4; ++mt) {
            const _Float16* p = &lA[(waveM * 64 + mt * 16 + lrow) * LDSS + lhi * 8];
            v8h lo = *(const v8h*)p;
            v8h hi = *(const v8h*)(p + 16);
            af[mt] = __builtin_shufflevector(lo, hi, 0,1,2,3,4,5,6,7,8,9,10,11,12,13,14,15);
        }
        // B 32x16: lane holds column n=lrow; lanes<16 K {0..7, 16..23}; lanes>=16 K {8..15, 24..31}
        #pragma unroll
        for (int nt = 0; nt < 4; ++nt) {
            const _Float16* p = &lB[(waveN * 64 + nt * 16 + lrow) * LDSS + lhi * 8];
            v8h lo = *(const v8h*)p;
            v8h hi = *(const v8h*)(p + 16);
            v16h bf = __builtin_shufflevector(lo, hi, 0,1,2,3,4,5,6,7,8,9,10,11,12,13,14,15);
            #pragma unroll
            for (int mt = 0; mt < 4; ++mt)
                acc[mt][nt] = __builtin_amdgcn_wmma_f32_16x16x32_f16(
                    false, af[mt], false, bf, (short)0, acc[mt][nt], false, false);
        }
    }

    // Epilogue: C/D layout -> VGPR r, lanes<16: (M=r, N=lrow); lanes>=16: (M=8+r, N=lrow)
    #pragma unroll
    for (int nt = 0; nt < 4; ++nt) {
        const int col = tileN + waveN * 64 + nt * 16 + lrow;
        const float bs = bSq[col];
        #pragma unroll
        for (int mt = 0; mt < 4; ++mt) {
            const int rowBase = tileM + waveM * 64 + mt * 16 + lhi * 8;
            #pragma unroll
            for (int r = 0; r < 8; ++r) {
                const int row = rowBase + r;
                float sq = aSq[row] + bs - 2.0f * acc[mt][nt][r];
                sq = fmaxf(sq, 0.0f);
                out[(size_t)row * NROWS + col] = __expf(-GAMMA * sq);
            }
        }
    }
}

// ---------------------------------------------------------------------------
extern "C" void kernel_launch(void* const* d_in, const int* in_sizes, int n_in,
                              void* d_out, int out_size, void* d_ws, size_t ws_size,
                              hipStream_t stream)
{
    (void)in_sizes; (void)n_in; (void)out_size; (void)ws_size;

    const float* a = (const float*)d_in[0];
    const float* b = (const float*)d_in[1];
    float* out = (float*)d_out;

    // Workspace layout: aH (4MB) | bH (4MB) | aSq (32KB) | bSq (32KB)
    _Float16* aH = (_Float16*)d_ws;
    _Float16* bH = aH + (size_t)NROWS * KDIM;
    float* aSq = (float*)(bH + (size_t)NROWS * KDIM);
    float* bSq = aSq + NROWS;

    rbf_prep<<<2 * NROWS, 256, 0, stream>>>(a, b, aH, bH, aSq, bSq);

    dim3 grid(NROWS / BN, NROWS / BM);   // 64 x 32 tiles
    rbf_wmma<<<grid, 256, 0, stream>>>(aH, bH, aSq, bSq, out);
}
